// MultiHeadAttention_84164179132604
// MI455X (gfx1250) — compile-verified, matrix operand order checked by reading, not run on hardware
//
#include <hip/hip_runtime.h>
#include <hip/hip_bf16.h>

// ---------------------------------------------------------------------------
// MI455X (gfx1250) fused multi-head attention, f16 WMMA with f32 accumulation.
// Pipeline: convert -> QKV GEMM (fused head split, Q pre-scaled, V transposed)
//           -> flash-attention (online softmax, LDS transpose of P)
//           -> output projection GEMM (+bias) -> f32 d_out.
// Fat per-wave tiles (4x4 / 4x2 WMMA tiles, 32-query attention tiles) to
// maximize the WMMA:b128-load ratio on the WGP-cache path.
// ---------------------------------------------------------------------------

typedef _Float16 h8_t  __attribute__((ext_vector_type(8)));
typedef _Float16 v16h  __attribute__((ext_vector_type(16)));
typedef float    v8f   __attribute__((ext_vector_type(8)));

#define DIM_C      1024
#define NSEQ_C     2048
#define NHEADS_C   16
#define HDIM_C     64
#define BATCH_C    2
#define ROWS_C     (BATCH_C * NSEQ_C)                       // 4096
#define QKV_ELEMS  (BATCH_C * NHEADS_C * NSEQ_C * HDIM_C)   // 4,194,304

// Load one 16-lane-group fragment: lane holds K = base+{0..7} and base+16+{0..7}.
// Caller passes  ptr = row_base + g*8 ; we grab [ptr, ptr+8) and [ptr+16, ptr+24).
__device__ __forceinline__ v16h load_frag(const _Float16* p) {
    h8_t lo = *(const h8_t*)(p);
    h8_t hi = *(const h8_t*)(p + 16);
    return __builtin_shufflevector(lo, hi, 0, 1, 2, 3, 4, 5, 6, 7,
                                           8, 9, 10, 11, 12, 13, 14, 15);
}

__device__ __forceinline__ v8f wmma_f16(v16h a, v16h b, v8f c) {
    // D = A(16x32 f16) * B(32x16 f16) + C(16x16 f32)
    return __builtin_amdgcn_wmma_f32_16x16x32_f16(
        /*neg_a=*/false, a, /*neg_b=*/false, b,
        /*c_mod=*/(short)0, c, /*reuse_a=*/false, /*reuse_b=*/false);
}

// ---------------------------------------------------------------------------
// f32 -> f16 elementwise convert, 8 elems/thread, 128-bit loads/stores.
// ---------------------------------------------------------------------------
__global__ void mha_cvt_f16(const float* __restrict__ src,
                            _Float16* __restrict__ dst, int n) {
    int i = (blockIdx.x * blockDim.x + threadIdx.x) * 8;
    if (i + 8 <= n) {
        const float4* s = (const float4*)(src + i);
        float4 a = s[0];
        float4 b = s[1];
        h8_t o;
        o[0] = (_Float16)a.x; o[1] = (_Float16)a.y;
        o[2] = (_Float16)a.z; o[3] = (_Float16)a.w;
        o[4] = (_Float16)b.x; o[5] = (_Float16)b.y;
        o[6] = (_Float16)b.z; o[7] = (_Float16)b.w;
        *(h8_t*)(dst + i) = o;
    }
}

// ---------------------------------------------------------------------------
// QKV GEMM: qkv[m, j] = sum_k xh[m,k] * Wh[j,k] + bias[j]
// One wave per block computes a 64x64 macro-tile (4x4 WMMA tiles):
// 16 WMMAs per 16 b128 loads per K-step.
// Epilogue scatters into Q [b,h,n,hd] (scaled 1/8), K [b,h,n,hd], Vt [b,h,hd,n].
// ---------------------------------------------------------------------------
__global__ __launch_bounds__(32)
void mha_gemm_qkv(const _Float16* __restrict__ A,   // [4096,1024]
                  const _Float16* __restrict__ W,   // [3072,1024]
                  const float*    __restrict__ bias,// [3072]
                  _Float16* __restrict__ Qd,
                  _Float16* __restrict__ Kd,
                  _Float16* __restrict__ Vtd) {
    const int lane = threadIdx.x;
    const int g = lane >> 4, r = lane & 15;
    const int n0 = blockIdx.x * 64;
    const int m0 = blockIdx.y * 64;

    v8f acc[4][4] = {};
    const _Float16* Ap = A + (size_t)(m0 + r) * DIM_C + g * 8;
    const _Float16* Wp = W + (size_t)(n0 + r) * DIM_C + g * 8;

    for (int k0 = 0; k0 < DIM_C; k0 += 32) {
        v16h af[4], bf[4];
#pragma unroll
        for (int mi = 0; mi < 4; ++mi)
            af[mi] = load_frag(Ap + (size_t)(mi * 16) * DIM_C + k0);
#pragma unroll
        for (int ni = 0; ni < 4; ++ni)
            bf[ni] = load_frag(Wp + (size_t)(ni * 16) * DIM_C + k0);
#pragma unroll
        for (int mi = 0; mi < 4; ++mi)
#pragma unroll
            for (int ni = 0; ni < 4; ++ni)
                acc[mi][ni] = wmma_f16(af[mi], bf[ni], acc[mi][ni]);
    }

#pragma unroll
    for (int ni = 0; ni < 4; ++ni) {
        const int j = n0 + ni * 16 + r;            // column in [0, 3072)
        const float bb = bias[j];
        const int which = j >> 10;                 // 0=Q 1=K 2=V
        const int d = j & 1023;
        const int hh = d >> 6;
        const int hdi = d & 63;
#pragma unroll
        for (int mi = 0; mi < 4; ++mi) {
#pragma unroll
            for (int i = 0; i < 8; ++i) {
                const int m = m0 + mi * 16 + g * 8 + i;   // global row
                const int bIdx = m >> 11;                 // batch
                const int nseq = m & 2047;                // seq position
                const float v = acc[mi][ni][i] + bb;
                if (which == 0) {
                    Qd[(((size_t)bIdx * NHEADS_C + hh) * NSEQ_C + nseq) * HDIM_C + hdi] =
                        (_Float16)(v * 0.125f);           // pre-scale by 1/sqrt(64)
                } else if (which == 1) {
                    Kd[(((size_t)bIdx * NHEADS_C + hh) * NSEQ_C + nseq) * HDIM_C + hdi] =
                        (_Float16)v;
                } else {
                    Vtd[(((size_t)bIdx * NHEADS_C + hh) * HDIM_C + hdi) * NSEQ_C + nseq] =
                        (_Float16)v;
                }
            }
        }
    }
}

// ---------------------------------------------------------------------------
// Fused attention: one wave per 32-query tile of one (b,h).
// Online softmax over key blocks of 32; P transposed C->A layout via LDS.
// K/V fragments are shared across the two query M-tiles.
// ---------------------------------------------------------------------------
__global__ __launch_bounds__(32)
void mha_flash_attn(const _Float16* __restrict__ Q,
                    const _Float16* __restrict__ K,
                    const _Float16* __restrict__ Vt,
                    _Float16* __restrict__ Oh) {      // [B,N,DIM] f16
    __shared__ __align__(16) _Float16 Plds[32 * 32];

    const int lane = threadIdx.x;
    const int g = lane >> 4, r = lane & 15;
    const int bh = blockIdx.y;
    const int b = bh >> 4, h = bh & 15;
    const int q0 = blockIdx.x * 32;

    const _Float16* Qp = Q + (size_t)bh * NSEQ_C * HDIM_C;
    const _Float16* Kp = K + (size_t)bh * NSEQ_C * HDIM_C;
    const _Float16* Vp = Vt + (size_t)bh * HDIM_C * NSEQ_C;

    // Q tiles (2 x 16 rows x 64) fragments, fixed for the whole block.
    v16h qf[2][2];
#pragma unroll
    for (int mi = 0; mi < 2; ++mi)
#pragma unroll
        for (int kk = 0; kk < 2; ++kk)
            qf[mi][kk] =
                load_frag(Qp + (size_t)(q0 + mi * 16 + r) * HDIM_C + kk * 32 + g * 8);

    v8f o[2][4] = {};
    float mrun[2][8], lrun[2][8];
#pragma unroll
    for (int mi = 0; mi < 2; ++mi)
#pragma unroll
        for (int i = 0; i < 8; ++i) { mrun[mi][i] = -3.0e38f; lrun[mi][i] = 0.0f; }

    for (int kb = 0; kb < NSEQ_C; kb += 32) {
        // ---- scores S = Q * K^T for 32 keys (two 16-wide N tiles) ----
        v16h kfr[2][2];
#pragma unroll
        for (int nt = 0; nt < 2; ++nt)
#pragma unroll
            for (int kk = 0; kk < 2; ++kk)
                kfr[nt][kk] =
                    load_frag(Kp + (size_t)(kb + nt * 16 + r) * HDIM_C + kk * 32 + g * 8);

        v8f s[2][2] = {};
#pragma unroll
        for (int mi = 0; mi < 2; ++mi)
#pragma unroll
            for (int nt = 0; nt < 2; ++nt) {
                s[mi][nt] = wmma_f16(qf[mi][0], kfr[nt][0], s[mi][nt]);
                s[mi][nt] = wmma_f16(qf[mi][1], kfr[nt][1], s[mi][nt]);
            }

        // ---- online softmax update (row = mi*16 + i + g*8, cols over lanes) ----
#pragma unroll
        for (int mi = 0; mi < 2; ++mi) {
#pragma unroll
            for (int i = 0; i < 8; ++i) {
                float mx = fmaxf(s[mi][0][i], s[mi][1][i]);
                mx = fmaxf(mx, __shfl_xor(mx, 1, 32));
                mx = fmaxf(mx, __shfl_xor(mx, 2, 32));
                mx = fmaxf(mx, __shfl_xor(mx, 4, 32));
                mx = fmaxf(mx, __shfl_xor(mx, 8, 32));
                const float mN = fmaxf(mrun[mi][i], mx);
                const float corr = __expf(mrun[mi][i] - mN);
                const float p0 = __expf(s[mi][0][i] - mN);
                const float p1 = __expf(s[mi][1][i] - mN);
                float rs = p0 + p1;
                rs += __shfl_xor(rs, 1, 32);
                rs += __shfl_xor(rs, 2, 32);
                rs += __shfl_xor(rs, 4, 32);
                rs += __shfl_xor(rs, 8, 32);
                lrun[mi][i] = lrun[mi][i] * corr + rs;
                mrun[mi][i] = mN;
#pragma unroll
                for (int t = 0; t < 4; ++t) o[mi][t][i] *= corr;
                Plds[(mi * 16 + g * 8 + i) * 32 + r]      = (_Float16)p0;
                Plds[(mi * 16 + g * 8 + i) * 32 + 16 + r] = (_Float16)p1;
            }
        }
        __syncthreads();

        // ---- P back in A layout; O += P (32x32) * V (32x64) ----
        v16h pf[2];
#pragma unroll
        for (int mi = 0; mi < 2; ++mi)
            pf[mi] = load_frag(&Plds[(mi * 16 + r) * 32 + g * 8]);
#pragma unroll
        for (int t = 0; t < 4; ++t) {
            v16h vfrag = load_frag(Vp + (size_t)(t * 16 + r) * NSEQ_C + kb + g * 8);
            o[0][t] = wmma_f16(pf[0], vfrag, o[0][t]);
            o[1][t] = wmma_f16(pf[1], vfrag, o[1][t]);
        }
        __syncthreads();
    }

    // ---- normalize and scatter to [B, N, H*hd] f16 ----
#pragma unroll
    for (int mi = 0; mi < 2; ++mi) {
#pragma unroll
        for (int i = 0; i < 8; ++i) {
            const float inv = 1.0f / lrun[mi][i];
            const int m = q0 + mi * 16 + g * 8 + i;
#pragma unroll
            for (int t = 0; t < 4; ++t) {
                Oh[((size_t)b * NSEQ_C + m) * DIM_C + h * HDIM_C + t * 16 + r] =
                    (_Float16)(o[mi][t][i] * inv);
            }
        }
    }
}

// ---------------------------------------------------------------------------
// Output projection: out[m,j] = sum_k Ah[m,k] * Wh[j,k] + bias[j]  (f32 out)
// One wave per block computes a 64x32 macro-tile (4x2 WMMA tiles).
// ---------------------------------------------------------------------------
__global__ __launch_bounds__(32)
void mha_gemm_out(const _Float16* __restrict__ A,   // [4096,1024]
                  const _Float16* __restrict__ W,   // [1024,1024]
                  const float*    __restrict__ bias,// [1024]
                  float* __restrict__ out) {        // [4096,1024]
    const int lane = threadIdx.x;
    const int g = lane >> 4, r = lane & 15;
    const int n0 = blockIdx.x * 32;
    const int m0 = blockIdx.y * 64;

    v8f acc[4][2] = {};
    const _Float16* Ap = A + (size_t)(m0 + r) * DIM_C + g * 8;
    const _Float16* Wp = W + (size_t)(n0 + r) * DIM_C + g * 8;

    for (int k0 = 0; k0 < DIM_C; k0 += 32) {
        v16h af[4], bf[2];
#pragma unroll
        for (int mi = 0; mi < 4; ++mi)
            af[mi] = load_frag(Ap + (size_t)(mi * 16) * DIM_C + k0);
#pragma unroll
        for (int ni = 0; ni < 2; ++ni)
            bf[ni] = load_frag(Wp + (size_t)(ni * 16) * DIM_C + k0);
#pragma unroll
        for (int mi = 0; mi < 4; ++mi)
#pragma unroll
            for (int ni = 0; ni < 2; ++ni)
                acc[mi][ni] = wmma_f16(af[mi], bf[ni], acc[mi][ni]);
    }

#pragma unroll
    for (int ni = 0; ni < 2; ++ni) {
        const int j = n0 + ni * 16 + r;
        const float bb = bias[j];
#pragma unroll
        for (int mi = 0; mi < 4; ++mi) {
#pragma unroll
            for (int i = 0; i < 8; ++i) {
                const int m = m0 + mi * 16 + g * 8 + i;
                out[(size_t)m * DIM_C + j] = acc[mi][ni][i] + bb;
            }
        }
    }
}

// ---------------------------------------------------------------------------
// Host launcher
// ---------------------------------------------------------------------------
extern "C" void kernel_launch(void* const* d_in, const int* in_sizes, int n_in,
                              void* d_out, int out_size, void* d_ws, size_t ws_size,
                              hipStream_t stream) {
    const float* x      = (const float*)d_in[0];   // [2,2048,1024]
    const float* qkv_w  = (const float*)d_in[1];   // [3072,1024]
    const float* qkv_b  = (const float*)d_in[2];   // [3072]
    const float* out_w  = (const float*)d_in[3];   // [1024,1024]
    const float* out_b  = (const float*)d_in[4];   // [1024]
    float* out = (float*)d_out;

    const size_t XH  = (size_t)ROWS_C * DIM_C;       // 4,194,304
    const size_t QW  = (size_t)3 * DIM_C * DIM_C;    // 3,145,728
    const size_t OW  = (size_t)DIM_C * DIM_C;        // 1,048,576
    const size_t QKV = (size_t)QKV_ELEMS;            // 4,194,304
    const size_t AH  = (size_t)ROWS_C * DIM_C;       // 4,194,304

    char* ws = (char*)d_ws;
    size_t off = 0;
    auto take = [&](size_t elems) {
        _Float16* p = (_Float16*)(ws + off);
        off += (elems * sizeof(_Float16) + 255) & ~(size_t)255;
        return p;
    };
    _Float16* xh  = take(XH);
    _Float16* qwh = take(QW);
    _Float16* owh = take(OW);
    _Float16* Qd  = take(QKV);
    _Float16* Kd  = take(QKV);
    _Float16* Vtd = take(QKV);
    _Float16* ah  = take(AH);
    (void)ws_size; (void)in_sizes; (void)n_in; (void)out_size;

    // 1) f32 -> f16 conversions (all sizes are multiples of 2048)
    mha_cvt_f16<<<dim3((unsigned)(XH / 2048)), 256, 0, stream>>>(x, xh, (int)XH);
    mha_cvt_f16<<<dim3((unsigned)(QW / 2048)), 256, 0, stream>>>(qkv_w, qwh, (int)QW);
    mha_cvt_f16<<<dim3((unsigned)(OW / 2048)), 256, 0, stream>>>(out_w, owh, (int)OW);

    // 2) QKV projection + head split (Q pre-scaled, V transposed): 64x64 tiles
    mha_gemm_qkv<<<dim3(3 * DIM_C / 64, ROWS_C / 64), 32, 0, stream>>>(
        xh, qwh, qkv_b, Qd, Kd, Vtd);

    // 3) fused flash attention: grid = (32-query tiles, B*H)
    mha_flash_attn<<<dim3(NSEQ_C / 32, BATCH_C * NHEADS_C), 32, 0, stream>>>(
        Qd, Kd, Vtd, ah);

    // 4) output projection + bias -> f32: 64x32 tiles
    mha_gemm_out<<<dim3(DIM_C / 32, ROWS_C / 64), 32, 0, stream>>>(
        ah, owh, out_b, out);
}
